// MeshPool_54047868453470
// MI455X (gfx1250) — compile-verified, matrix operand order checked by reading, not run on hardware
//
#include <hip/hip_runtime.h>
#include <hip/hip_bf16.h>

typedef __attribute__((ext_vector_type(16))) _Float16 v16h;
typedef __attribute__((ext_vector_type(8)))  float    v8f;
typedef __attribute__((ext_vector_type(4)))  float    v4f;
typedef unsigned long long u64;

// ---------------------------------------------------------------------------
// K1: per-vertex squared norms via v_wmma_f32_16x16x32_f16.
// One wave32 per 16-row tile.  A = elementwise squares (f16), B = ones,
// C/D = f32 accumulator -> column 0 of D holds the 16 row sums.
// With B == ones the WMMA A-layout K-permutation is irrelevant: any
// arrangement of the 32 squared values across lane L / lane L+16 sums the
// same, so each lane just loads 16 contiguous floats of its row.
// ---------------------------------------------------------------------------
__global__ __launch_bounds__(256) void vnorm_wmma(const float* __restrict__ image,
                                                  float* __restrict__ vnorm,
                                                  int V, int F) {
  int wave  = (int)((blockIdx.x * blockDim.x + threadIdx.x) >> 5);
  int lane  = (int)(threadIdx.x & 31);
  int ntile = (V + 15) >> 4;
  if (wave >= ntile) return;                  // wave-uniform: EXEC stays all-1s
  int row  = wave * 16 + (lane & 15);
  int rowc = row < V ? row : V - 1;           // clamp (dup read) if V % 16 != 0
  int half = lane >> 4;

  v16h ones;
  #pragma unroll
  for (int i = 0; i < 16; ++i) ones[i] = (_Float16)1.0f;

  v8f acc = {};
  const float* rp = image + (size_t)rowc * F + half * 16;
  int nc = F >> 5;                            // K chunks of 32
  for (int c = 0; c < nc; ++c) {
    const v4f* p = (const v4f*)(rp + c * 32);
    v4f x0 = p[0], x1 = p[1], x2 = p[2], x3 = p[3];
    v16h a;
    #pragma unroll
    for (int i = 0; i < 4; ++i) {
      a[i]      = (_Float16)(x0[i] * x0[i]);
      a[i + 4]  = (_Float16)(x1[i] * x1[i]);
      a[i + 8]  = (_Float16)(x2[i] * x2[i]);
      a[i + 12] = (_Float16)(x3[i] * x3[i]);
    }
    // (neg_a, A, neg_b, B, c_mod, C, reuse_a, reuse_b)
    acc = __builtin_amdgcn_wmma_f32_16x16x32_f16(false, a, false, ones,
                                                 (short)0, acc, false, false);
  }

  // D layout: lanes 0..15 -> N=lane, VGPR r -> M=r; lanes 16..31 -> M=8+r.
  // Column 0 lives in lane 0 (rows 0..7) and lane 16 (rows 8..15).
  if ((lane & 15) == 0) {
    int mbase = wave * 16 + half * 8;
    #pragma unroll
    for (int r = 0; r < 8; ++r) {
      int m = mbase + r;
      if (m < V) vnorm[m] = acc[r];
    }
  }
}

// ---------------------------------------------------------------------------
// K2: per-edge priority, boundary flag, and 64-bit sort key.
// key = (orderable(sq) << 32) | edge_id  -> ascending u64 order ==
// ascending float with stable index tie-break (matches heapify [value,id]).
// Pads [E, N) with max keys every call.
// ---------------------------------------------------------------------------
__global__ void edge_keys(const float* __restrict__ vnorm,
                          const float* __restrict__ vs,
                          const int*   __restrict__ edges,
                          float* __restrict__ out_sq,
                          u64*   __restrict__ keys,
                          int*   __restrict__ bnd,
                          int E, unsigned N, float eps) {
  unsigned e = blockIdx.x * blockDim.x + threadIdx.x;
  if (e >= N) return;
  if (e >= (unsigned)E) { keys[e] = ~0ull; return; }
  int v0 = edges[e], v1 = edges[E + e];
  float sq = vnorm[v0] + vnorm[v1];
  out_sq[e] = sq;
  float ax = vs[2 * v0], ay = vs[2 * v0 + 1];
  float bx = vs[2 * v1], by = vs[2 * v1 + 1];
  float hi = 1.0f - eps;
  int b = (ax < eps) | (ax > hi) | (ay < eps) | (ay > hi) |
          (bx < eps) | (bx > hi) | (by < eps) | (by > hi);
  bnd[e] = b;
  unsigned u = __float_as_uint(sq);
  u = (u & 0x80000000u) ? ~u : (u | 0x80000000u);
  keys[e] = ((u64)u << 32) | (u64)e;
}

// ---------------------------------------------------------------------------
// Bitonic sort (L2-resident: 2 MB of keys).  Direction from global index.
// ---------------------------------------------------------------------------
__device__ __forceinline__ void bitonic_ce(u64& a, u64& b, bool up) {
  if ((up && a > b) || (!up && a < b)) { u64 t = a; a = b; b = t; }
}

// All stages k = 2..1024 inside one 1024-element LDS chunk.
__global__ __launch_bounds__(512) void bitonic_local_full(u64* __restrict__ keys) {
  __shared__ u64 s[1024];
  unsigned t = threadIdx.x, base = blockIdx.x * 1024u;
  s[t] = keys[base + t];
  s[t + 512] = keys[base + t + 512];
  __syncthreads();
  for (unsigned k = 2; k <= 1024; k <<= 1) {
    for (unsigned j = k >> 1; j > 0; j >>= 1) {
      unsigned i = ((t & ~(j - 1)) << 1) | (t & (j - 1));
      unsigned l = i | j;
      u64 a = s[i], b = s[l];
      bitonic_ce(a, b, ((base + i) & k) == 0);
      s[i] = a; s[l] = b;
      __syncthreads();
    }
  }
  keys[base + t] = s[t];
  keys[base + t + 512] = s[t + 512];
}

// One global compare-exchange step (j >= 1024).
__global__ __launch_bounds__(512) void bitonic_global_step(u64* __restrict__ keys,
                                                           unsigned j, unsigned k) {
  unsigned i = blockIdx.x * blockDim.x + threadIdx.x;
  unsigned l = i ^ j;
  if (l > i) {
    u64 a = keys[i], b = keys[l];
    bool up = (i & k) == 0;
    if ((up && a > b) || (!up && a < b)) { keys[i] = b; keys[l] = a; }
  }
}

// Remaining steps j = 512..1 of stage k, fused in LDS.
__global__ __launch_bounds__(512) void bitonic_local_steps(u64* __restrict__ keys,
                                                           unsigned k) {
  __shared__ u64 s[1024];
  unsigned t = threadIdx.x, base = blockIdx.x * 1024u;
  s[t] = keys[base + t];
  s[t + 512] = keys[base + t + 512];
  __syncthreads();
  for (unsigned j = 512; j > 0; j >>= 1) {
    unsigned i = ((t & ~(j - 1)) << 1) | (t & (j - 1));
    unsigned l = i | j;
    u64 a = s[i], b = s[l];
    bitonic_ce(a, b, ((base + i) & k) == 0);
    s[i] = a; s[l] = b;
    __syncthreads();
  }
  keys[base + t] = s[t];
  keys[base + t + 512] = s[t + 512];
}

// ---------------------------------------------------------------------------
// K4: gather sorted edge tuples into contiguous arrays (sequential reads for
// the serial pass).
// ---------------------------------------------------------------------------
__global__ void gather_sorted(const u64* __restrict__ keys,
                              const int* __restrict__ edges,
                              const int* __restrict__ bnd,
                              int* __restrict__ e0s, int* __restrict__ e1s,
                              int* __restrict__ bs, int E) {
  int p = (int)(blockIdx.x * blockDim.x + threadIdx.x);
  if (p >= E) return;
  unsigned e = (unsigned)(keys[p] & 0xFFFFFFFFull);
  e0s[p] = edges[e];
  e1s[p] = edges[E + e];
  bs[p]  = bnd[e];
}

// ---------------------------------------------------------------------------
// K5: sequential greedy collapse.  Alive mask (V bytes) lives in LDS
// (fits easily in the 320 KB WGP LDS).  Early exit: once count == target the
// reference's `do` predicate is false forever, so the mask is final.
// ---------------------------------------------------------------------------
__global__ __launch_bounds__(256) void greedy_collapse(const int* __restrict__ e0s,
                                                       const int* __restrict__ e1s,
                                                       const int* __restrict__ bs,
                                                       const int* __restrict__ target_p,
                                                       float* __restrict__ out_mask,
                                                       float* __restrict__ out_count,
                                                       int V, int E) {
  extern __shared__ unsigned char mask[];
  __shared__ int s_count;
  for (int v = (int)threadIdx.x; v < V; v += (int)blockDim.x) mask[v] = 1;
  __syncthreads();
  if (threadIdx.x == 0) {
    int target = *target_p;
    int count  = V;
    for (int p = 0; p < E; ++p) {
      if (count <= target) break;
      if ((p & 63) == 0) {                 // stream next tuples into L2/L0
        __builtin_prefetch(e0s + p + 256, 0, 0);
        __builtin_prefetch(e1s + p + 256, 0, 0);
        __builtin_prefetch(bs  + p + 256, 0, 0);
      }
      int v1 = e1s[p];
      if (!bs[p] && mask[e0s[p]] && mask[v1]) { mask[v1] = 0; --count; }
    }
    s_count = count;
  }
  __syncthreads();
  for (int v = (int)threadIdx.x; v < V; v += (int)blockDim.x)
    out_mask[v] = (float)mask[v];
  if (threadIdx.x == 0) *out_count = (float)s_count;
}

// ---------------------------------------------------------------------------
extern "C" void kernel_launch(void* const* d_in, const int* in_sizes, int n_in,
                              void* d_out, int out_size, void* d_ws, size_t ws_size,
                              hipStream_t stream) {
  const float* image = (const float*)d_in[0];
  const float* vs    = (const float*)d_in[1];
  const int*   edges = (const int*)d_in[2];
  const int*   tgt   = (const int*)d_in[3];

  int V = in_sizes[1] / 2;          // vs is [V,2]
  int E = in_sizes[2] / 2;          // edges is [2,E]
  int F = in_sizes[0] / V;          // image is [V,F]

  unsigned N = 2048;                // sort size: pow2 >= E, >= 2048
  while (N < (unsigned)E) N <<= 1;

  char* ws = (char*)d_ws;
  size_t off = 0;
  u64*   keys  = (u64*)(ws + off);  off += (size_t)N * 8;
  float* vnorm = (float*)(ws + off); off += (((size_t)V * 4) + 255) & ~(size_t)255;
  int*   bnd   = (int*)(ws + off);   off += (((size_t)E * 4) + 255) & ~(size_t)255;
  int*   e0s   = (int*)(ws + off);   off += (((size_t)E * 4) + 255) & ~(size_t)255;
  int*   e1s   = (int*)(ws + off);   off += (((size_t)E * 4) + 255) & ~(size_t)255;
  int*   bs    = (int*)(ws + off);   off += (((size_t)E * 4) + 255) & ~(size_t)255;
  (void)ws_size;

  float* out_sq    = (float*)d_out;
  float* out_mask  = out_sq + E;
  float* out_count = out_mask + V;
  (void)out_size;

  // K1: vertex norms (8 waves / block, one 16-row tile per wave)
  int ntile  = (V + 15) / 16;
  int nblk1  = (ntile + 7) / 8;
  vnorm_wmma<<<nblk1, 256, 0, stream>>>(image, vnorm, V, F);

  // K2: keys + boundary + sq output (+ pad keys)
  edge_keys<<<(N + 255) / 256, 256, 0, stream>>>(vnorm, vs, edges, out_sq, keys,
                                                 bnd, E, N, 0.01f);

  // K3: bitonic sort of N u64 keys
  unsigned chunks = N / 1024;
  bitonic_local_full<<<chunks, 512, 0, stream>>>(keys);
  for (unsigned k = 2048; k <= N; k <<= 1) {
    for (unsigned j = k >> 1; j >= 1024; j >>= 1)
      bitonic_global_step<<<N / 512, 512, 0, stream>>>(keys, j, k);
    bitonic_local_steps<<<chunks, 512, 0, stream>>>(keys, k);
  }

  // K4: gather sorted tuples
  gather_sorted<<<(E + 255) / 256, 256, 0, stream>>>(keys, edges, bnd, e0s, e1s,
                                                     bs, E);

  // K5: sequential greedy collapse (single block, mask in LDS)
  size_t shbytes = (size_t)V + 64;
  greedy_collapse<<<1, 256, shbytes, stream>>>(e0s, e1s, bs, tgt, out_mask,
                                               out_count, V, E);
}
